// TransformerBlock_73040213836212
// MI455X (gfx1250) — matrix operand order repaired and checked by reading, not compiled, not run on hardware
//
#include <hip/hip_runtime.h>
#include <hip/hip_bf16.h>
#include <math.h>

// ---------------------------------------------------------------------------
// MI455X / gfx1250 transformer block with future-attention.
// GEMMs: bf16 V_WMMA_F32_16X16X32_BF16, LDS-tiled (128x128x32 macro tiles),
// double-buffered staging via TDM (tensor_load_to_lds) or async-to-LDS loads.
// ---------------------------------------------------------------------------

typedef __attribute__((ext_vector_type(16))) __bf16 v16bf;
typedef __attribute__((ext_vector_type(8)))  __bf16 v8bf;
typedef __attribute__((ext_vector_type(8)))  float  v8f;
typedef __attribute__((ext_vector_type(4)))  unsigned int v4u;
typedef __attribute__((ext_vector_type(8)))  int v8i;
typedef __attribute__((ext_vector_type(4)))  int v4i;

#define Dm   1024
#define Tm   1024
#define Bm   2
#define Hm   16
#define HSm  64
#define Fm   4
#define BTm  (Bm * Tm)

#define GLOBAL_AS __attribute__((address_space(1)))
#define LDS_AS    __attribute__((address_space(3)))

#ifndef __has_builtin
#define __has_builtin(x) 0
#endif

#if __has_builtin(__builtin_amdgcn_tensor_load_to_lds) && \
    __has_builtin(__builtin_amdgcn_s_wait_tensorcnt)
#define USE_TDM 1
#else
#define USE_TDM 0
#endif
#if __has_builtin(__builtin_amdgcn_global_load_async_to_lds_b128)
#define USE_ASYNC 1
#else
#define USE_ASYNC 0
#endif

__device__ __forceinline__ v8f wmma_bf16(v16bf a, v16bf b, v8f c) {
  return __builtin_amdgcn_wmma_f32_16x16x32_bf16(
      false, a, false, b, (short)0, c, false, false);
}

// A fragment (16x32): lane l -> row l&15, K = half*8+{0..7} and 16+half*8+{0..7}
template <typename PT>
__device__ __forceinline__ v16bf ld_afrag(PT tile, int ld, int k, int lane) {
  int hf = (lane >> 4) & 1;
  int r  = lane & 15;
  auto p = tile + (size_t)r * ld + k + hf * 8;
  v8bf lo = *(const v8bf*)(p);
  v8bf hi = *(const v8bf*)(p + 16);
  v16bf out;
#pragma unroll
  for (int i = 0; i < 8; ++i) { out[i] = lo[i]; out[i + 8] = hi[i]; }
  return out;
}

// B fragment (32x16) from (N x K) row-major: lane l -> col l&15, K = half*8+{0..7} and 16+half*8+{0..7}
template <typename PT>
__device__ __forceinline__ v16bf ld_bfrag(PT tile, int ld, int k, int lane) {
  int hf = (lane >> 4) & 1;
  int nn = lane & 15;
  auto p = tile + (size_t)nn * ld + k + hf * 8;
  v8bf lo = *(const v8bf*)(p);
  v8bf hi = *(const v8bf*)(p + 16);
  v16bf out;
#pragma unroll
  for (int i = 0; i < 8; ++i) { out[i] = lo[i]; out[i + 8] = hi[i]; }
  return out;
}

#if USE_TDM
__device__ __forceinline__ unsigned lds_off(const void* p) {
  return (unsigned)(unsigned long long)(LDS_AS const void*)p;
}
// TDM 2D tile load: tile = 32 elems (K) x 128 rows, bf16, row stride = ld elems.
__device__ __forceinline__ void tdm_stage_tile(const __bf16* gsrc, const __bf16* ldst,
                                               unsigned ld) {
  unsigned long long ga = (unsigned long long)(const void*)gsrc;
  v4u g0;
  g0[0] = 1u;                                              // count=1
  g0[1] = lds_off(ldst);                                   // lds_addr
  g0[2] = (unsigned)(ga & 0xffffffffull);                  // global_addr[31:0]
  g0[3] = (unsigned)((ga >> 32) & 0x01ffffffull) | 0x80000000u;  // [56:32] | type=2
  unsigned td0 = ld;            // tensor_dim0 (elements per row)
  unsigned td1 = 1u << 20;      // tensor_dim1 (plenty of rows)
  unsigned tile0 = 32u, tile1 = 128u;
  v8i g1;
  g1[0] = (int)(1u << 16);                                  // data_size=1 (2 bytes)
  g1[1] = (int)((td0 & 0xffffu) << 16);                     // tensor_dim0[15:0]
  g1[2] = (int)(((td0 >> 16) & 0xffffu) | ((td1 & 0xffffu) << 16));
  g1[3] = (int)(((td1 >> 16) & 0xffffu) | (tile0 << 16));   // | tile_dim0
  g1[4] = (int)(tile1 & 0xffffu);                           // tile_dim1, tile_dim2=0
  g1[5] = (int)ld;                                          // dim0_stride[31:0]
  g1[6] = 0;                                                // dim0_stride[47:32]=0
  g1[7] = 0;
  v4i gz = {0, 0, 0, 0};
#if __clang_major__ >= 23
  v8i gz8 = {0, 0, 0, 0, 0, 0, 0, 0};
  __builtin_amdgcn_tensor_load_to_lds(g0, g1, gz, gz, gz8, 0);
#else
  __builtin_amdgcn_tensor_load_to_lds(g0, g1, gz, gz, 0);
#endif
}
#endif

// ---------------------------------------------------------------------------
// fp32 -> bf16 conversion (+ transposed variant)
// ---------------------------------------------------------------------------
__global__ void cvt_bf16_kernel(const float* __restrict__ src,
                                __bf16* __restrict__ dst, int n) {
  for (int i = blockIdx.x * 256 + threadIdx.x; i < n; i += gridDim.x * 256)
    dst[i] = (__bf16)src[i];
}

__global__ void cvt_t_bf16_kernel(const float* __restrict__ src,
                                  __bf16* __restrict__ dst, int R, int C) {
  int tot = R * C;
  for (int i = blockIdx.x * 256 + threadIdx.x; i < tot; i += gridDim.x * 256) {
    int r = i / C, c = i - r * C;
    dst[(size_t)c * R + r] = (__bf16)src[i];
  }
}

// ---------------------------------------------------------------------------
// LayerNorm over D=1024
// ---------------------------------------------------------------------------
__global__ __launch_bounds__(256)
void ln_bf16_kernel(const float* __restrict__ x, const float* __restrict__ g,
                    const float* __restrict__ bta, __bf16* __restrict__ out) {
  int row = blockIdx.x;
  const float* xr = x + (size_t)row * Dm;
  __shared__ float s1[256], s2[256];
  float a = 0.f, b = 0.f;
  for (int i = threadIdx.x; i < Dm; i += 256) { float v = xr[i]; a += v; b += v * v; }
  s1[threadIdx.x] = a; s2[threadIdx.x] = b;
  __syncthreads();
  for (int off = 128; off > 0; off >>= 1) {
    if (threadIdx.x < off) {
      s1[threadIdx.x] += s1[threadIdx.x + off];
      s2[threadIdx.x] += s2[threadIdx.x + off];
    }
    __syncthreads();
  }
  float mu   = s1[0] * (1.f / Dm);
  float var  = s2[0] * (1.f / Dm) - mu * mu;
  float rstd = rsqrtf(var + 1e-5f);
  for (int i = threadIdx.x; i < Dm; i += 256)
    out[(size_t)row * Dm + i] = (__bf16)((xr[i] - mu) * rstd * g[i] + bta[i]);
}

// ---------------------------------------------------------------------------
// LDS-tiled bf16 WMMA GEMM:  C[M,N] = A[M,K] * W[N,K]^T  (+resid, +gelu)
// Block = 256 threads (8 waves). Macro tile 128(M) x 128(N), K-slab 32.
// Wave (wr,wc) computes 32x64: 2 row-tiles x 4 col-tiles (8 WMMAs / K-slab).
// A/B slabs double-buffered in LDS, staged by TDM / async-to-LDS loads.
// grid = (N/128, M/128)
// ---------------------------------------------------------------------------
__global__ __launch_bounds__(256)
void gemm_wmma_kernel(const __bf16* __restrict__ A, int lda,
                      const __bf16* __restrict__ Bw, int ldb,
                      int K,
                      const float* __restrict__ resid,
                      float* __restrict__ outF,
                      __bf16* __restrict__ outB,
                      int ldo, int act) {
  __shared__ __align__(16) __bf16 As[2][128 * 32];
  __shared__ __align__(16) __bf16 Bs[2][128 * 32];

  int tid  = threadIdx.x;
  int lane = tid & 31;
  int wave = tid >> 5;
  int wr = wave & 3;        // 4 row groups  -> rows wr*32 .. +31
  int wc = wave >> 2;       // 2 col groups  -> cols wc*64 .. +63
  int row0 = blockIdx.y * 128;
  int col0 = blockIdx.x * 128;
  const __bf16* Abase = A  + (size_t)row0 * lda;
  const __bf16* Bbase = Bw + (size_t)col0 * ldb;

  v8f acc[2][4];
#pragma unroll
  for (int i = 0; i < 2; ++i)
#pragma unroll
    for (int j = 0; j < 4; ++j) acc[i][j] = (v8f){};

  // ---- staging helpers --------------------------------------------------
  auto stage = [&](int k, int buf) {
#if USE_TDM
    if (wave == 0) {
      tdm_stage_tile(Abase + k, &As[buf][0], (unsigned)lda);
      tdm_stage_tile(Bbase + k, &Bs[buf][0], (unsigned)ldb);
    }
#else
    // 128 rows x 32 bf16 = 512 x 16B chunks per tile; 2 chunks/thread/tile
#pragma unroll
    for (int it = 0; it < 2; ++it) {
      int cid = tid + it * 256;
      int r = cid >> 2, ch = (cid & 3) * 8;
      const __bf16* ga = Abase + (size_t)r * lda + k + ch;
      const __bf16* gb = Bbase + (size_t)r * ldb + k + ch;
      __bf16* la = &As[buf][r * 32 + ch];
      __bf16* lb = &Bs[buf][r * 32 + ch];
#if USE_ASYNC
      __builtin_amdgcn_global_load_async_to_lds_b128(
          (const GLOBAL_AS void*)(const void*)ga, (LDS_AS void*)(void*)la, 0, 0);
      __builtin_amdgcn_global_load_async_to_lds_b128(
          (const GLOBAL_AS void*)(const void*)gb, (LDS_AS void*)(void*)lb, 0, 0);
#else
      *(v8bf*)la = *(const v8bf*)ga;
      *(v8bf*)lb = *(const v8bf*)gb;
#endif
    }
#endif
  };
  auto stage_wait = [&]() {
#if USE_TDM
    if (wave == 0) __builtin_amdgcn_s_wait_tensorcnt((short)0);
#elif USE_ASYNC
#if __has_builtin(__builtin_amdgcn_s_wait_asynccnt)
    __builtin_amdgcn_s_wait_asynccnt((short)0);
#else
    asm volatile("s_wait_asynccnt 0x0" ::: "memory");
#endif
#endif
  };

  // ---- main loop: double-buffered K slabs of 32 -------------------------
  int buf = 0;
  stage(0, buf);
  for (int k = 0; k < K; k += 32) {
    stage_wait();
    __syncthreads();
    if (k + 32 < K) stage(k + 32, buf ^ 1);

    v16bf af[2], bf[4];
#pragma unroll
    for (int rt = 0; rt < 2; ++rt)
      af[rt] = ld_afrag(&As[buf][(wr * 32 + rt * 16) * 32], 32, 0, lane);
#pragma unroll
    for (int ct = 0; ct < 4; ++ct)
      bf[ct] = ld_bfrag(&Bs[buf][(wc * 64 + ct * 16) * 32], 32, 0, lane);
#pragma unroll
    for (int rt = 0; rt < 2; ++rt)
#pragma unroll
      for (int ct = 0; ct < 4; ++ct)
        acc[rt][ct] = wmma_bf16(af[rt], bf[ct], acc[rt][ct]);

    __syncthreads();
    buf ^= 1;
  }

  // ---- epilogue ----------------------------------------------------------
  int nn = lane & 15, hf = lane >> 4;
#pragma unroll
  for (int rt = 0; rt < 2; ++rt) {
#pragma unroll
    for (int ct = 0; ct < 4; ++ct) {
#pragma unroll
      for (int i = 0; i < 8; ++i) {
        int r = row0 + wr * 32 + rt * 16 + hf * 8 + i;
        int c = col0 + wc * 64 + ct * 16 + nn;
        size_t idx = (size_t)r * ldo + c;
        float v = acc[rt][ct][i];
        if (resid) v += resid[idx];
        if (act)   v = 0.5f * v * (1.f + erff(v * 0.70710678118f));  // exact GELU
        if (outF)  outF[idx] = v;
        if (outB)  outB[idx] = (__bf16)v;
      }
    }
  }
}

// ---------------------------------------------------------------------------
// up_future: up[t, d', f] = sum_c f_head[t, c] * W_conv[c, d', f]  (per head)
// ---------------------------------------------------------------------------
__global__ __launch_bounds__(32)
void upfuture_kernel(const __bf16* __restrict__ qkvf,
                     const __bf16* __restrict__ Wct,   // (d'*4+f) x 64 row-major
                     __bf16* __restrict__ up) {        // (B,T,F,D)
  int lane = threadIdx.x;
  int t0 = blockIdx.x * 16, h = blockIdx.y, b = blockIdx.z;
  const __bf16* Fb = qkvf + ((size_t)(b * Tm + t0)) * (4 * Dm) + 3 * Dm + h * HSm;
  v16bf a0 = ld_afrag(Fb, 4 * Dm, 0, lane);
  v16bf a1 = ld_afrag(Fb, 4 * Dm, 32, lane);
  int nn = lane & 15, hf = lane >> 4;
  for (int nt = 0; nt < 16; ++nt) {
    const __bf16* Bt = Wct + (size_t)(nt * 16) * HSm;
    v8f c = {};
    c = wmma_bf16(a0, ld_bfrag(Bt, HSm, 0, lane), c);
    c = wmma_bf16(a1, ld_bfrag(Bt, HSm, 32, lane), c);
#pragma unroll
    for (int i = 0; i < 8; ++i) {
      int r = hf * 8 + i;
      int ng = nt * 16 + nn;      // = d'*4 + f
      int dp = ng >> 2, f = ng & 3;
      int t = t0 + r;
      up[(((size_t)(b * Tm + t)) * Fm + f) * Dm + h * HSm + dp] = (__bf16)c[i];
    }
  }
}

// ---------------------------------------------------------------------------
// Flash-style causal attention + future logits. One wave per (b, h, 16 rows).
// ---------------------------------------------------------------------------
__global__ __launch_bounds__(32)
void attn_kernel(const __bf16* __restrict__ qkvf,   // (B*T, 4D) q|k|v|f
                 const __bf16* __restrict__ kvf,    // (B*T*F, 2D) k_fut|v_fut
                 __bf16* __restrict__ out) {        // (B*T, D)
  __shared__ __align__(16) __bf16 plds[16 * 32];
  __shared__ __align__(16) __bf16 vt[64 * 32];
  int lane = threadIdx.x;
  int t0 = blockIdx.x * 16, h = blockIdx.y, b = blockIdx.z;
  int nn = lane & 15, hf = lane >> 4;

  const __bf16* Qb = qkvf + ((size_t)(b * Tm + t0)) * (4 * Dm) + h * HSm;
  v16bf qa0 = ld_afrag(Qb, 4 * Dm, 0, lane);
  v16bf qa1 = ld_afrag(Qb, 4 * Dm, 32, lane);

  float m16[16], l16[16];
#pragma unroll
  for (int r = 0; r < 16; ++r) { m16[r] = -3.0e38f; l16[r] = 0.f; }
  v8f O[4] = {{}, {}, {}, {}};

  int lastk = t0 + 15;
  for (int kb = 0; kb <= lastk; kb += 32) {
    v8f s[2];
#pragma unroll
    for (int j = 0; j < 2; ++j) {
      const __bf16* Kb = qkvf + ((size_t)(b * Tm + kb + j * 16)) * (4 * Dm) + Dm + h * HSm;
      v8f c = {};
      c = wmma_bf16(qa0, ld_bfrag(Kb, 4 * Dm, 0, lane), c);
      c = wmma_bf16(qa1, ld_bfrag(Kb, 4 * Dm, 32, lane), c);
#pragma unroll
      for (int i = 0; i < 8; ++i) {
        int rrow = t0 + hf * 8 + i;
        int col  = kb + j * 16 + nn;
        s[j][i] = (col <= rrow) ? c[i] * 0.125f : -3.0e38f;
      }
    }
    float bm[8];
#pragma unroll
    for (int i = 0; i < 8; ++i) bm[i] = fmaxf(s[0][i], s[1][i]);
#pragma unroll
    for (int i = 0; i < 8; ++i) {
      bm[i] = fmaxf(bm[i], __shfl_xor(bm[i], 1, 32));
      bm[i] = fmaxf(bm[i], __shfl_xor(bm[i], 2, 32));
      bm[i] = fmaxf(bm[i], __shfl_xor(bm[i], 4, 32));
      bm[i] = fmaxf(bm[i], __shfl_xor(bm[i], 8, 32));
    }
    float newm[16], alpha[16];
#pragma unroll
    for (int i = 0; i < 8; ++i) {
      int rme = hf * 8 + i, rot = (1 - hf) * 8 + i;
      float mo = __shfl_xor(bm[i], 16, 32);
      newm[rme] = fmaxf(m16[rme], bm[i]);
      newm[rot] = fmaxf(m16[rot], mo);
    }
#pragma unroll
    for (int r = 0; r < 16; ++r) alpha[r] = __expf(m16[r] - newm[r]);
    float psum[8];
#pragma unroll
    for (int i = 0; i < 8; ++i) {
      int r = hf * 8 + i;
      float p0 = __expf(s[0][i] - newm[r]);
      float p1 = __expf(s[1][i] - newm[r]);
      s[0][i] = p0; s[1][i] = p1; psum[i] = p0 + p1;
    }
#pragma unroll
    for (int i = 0; i < 8; ++i) {
      psum[i] += __shfl_xor(psum[i], 1, 32);
      psum[i] += __shfl_xor(psum[i], 2, 32);
      psum[i] += __shfl_xor(psum[i], 4, 32);
      psum[i] += __shfl_xor(psum[i], 8, 32);
    }
#pragma unroll
    for (int i = 0; i < 8; ++i) {
      int rme = hf * 8 + i, rot = (1 - hf) * 8 + i;
      float so = __shfl_xor(psum[i], 16, 32);
      l16[rme] = l16[rme] * alpha[rme] + psum[i];
      l16[rot] = l16[rot] * alpha[rot] + so;
      m16[rme] = newm[rme]; m16[rot] = newm[rot];
    }
#pragma unroll
    for (int i = 0; i < 8; ++i) {
      float a_ = alpha[hf * 8 + i];
      O[0][i] *= a_; O[1][i] *= a_; O[2][i] *= a_; O[3][i] *= a_;
    }
    __syncthreads();
#pragma unroll
    for (int i = 0; i < 8; ++i) {
      int r = hf * 8 + i;
      plds[r * 32 + nn]      = (__bf16)s[0][i];
      plds[r * 32 + 16 + nn] = (__bf16)s[1][i];
    }
    const __bf16* Vrow = qkvf + ((size_t)(b * Tm + kb + lane)) * (4 * Dm) + 2 * Dm + h * HSm;
#pragma unroll 4
    for (int d0 = 0; d0 < HSm; ++d0) vt[d0 * 32 + lane] = Vrow[d0];
    __syncthreads();
    v16bf pa = ld_afrag(plds, 32, 0, lane);
#pragma unroll
    for (int nt = 0; nt < 4; ++nt)
      O[nt] = wmma_bf16(pa, ld_bfrag(vt + (size_t)(nt * 16) * 32, 32, 0, lane), O[nt]);
  }

  // future logits: q[t] . k_future[t, f]  (unscaled, per reference)
  float flog[4] = {0.f, 0.f, 0.f, 0.f};
  if (lane < 16) {
    int t = t0 + lane;
    const __bf16* qrow = qkvf + ((size_t)(b * Tm + t)) * (4 * Dm) + h * HSm;
#pragma unroll
    for (int f = 0; f < Fm; ++f) {
      const __bf16* kf = kvf + (((size_t)(b * Tm + t)) * Fm + f) * (2 * Dm) + h * HSm;
      float acc = 0.f;
#pragma unroll 8
      for (int d = 0; d < HSm; ++d) acc += (float)qrow[d] * (float)kf[d];
      flog[f] = acc;
    }
  }
  float gm[16], beta[16], den[16];
#pragma unroll
  for (int r = 0; r < 16; ++r) {
    float f0 = __shfl(flog[0], r, 32), f1 = __shfl(flog[1], r, 32);
    float f2 = __shfl(flog[2], r, 32), f3 = __shfl(flog[3], r, 32);
    float fm = fmaxf(fmaxf(f0, f1), fmaxf(f2, f3));
    gm[r]   = fmaxf(m16[r], fm);
    beta[r] = __expf(m16[r] - gm[r]);
    den[r]  = l16[r] * beta[r] + __expf(f0 - gm[r]) + __expf(f1 - gm[r]) +
              __expf(f2 - gm[r]) + __expf(f3 - gm[r]);
  }
#pragma unroll
  for (int nt = 0; nt < 4; ++nt) {
#pragma unroll
    for (int i = 0; i < 8; ++i) {
      int r = hf * 8 + i, t = t0 + r, dim = nt * 16 + nn;
      float acc = O[nt][i] * beta[r];
#pragma unroll
      for (int f = 0; f < Fm; ++f) {
        float lg = __shfl(flog[f], r, 32);
        float vf = (float)kvf[(((size_t)(b * Tm + t)) * Fm + f) * (2 * Dm) + Dm + h * HSm + dim];
        acc += __expf(lg - gm[r]) * vf;
      }
      out[((size_t)(b * Tm + t)) * Dm + h * HSm + dim] = (__bf16)(acc / den[r]);
    }
  }
}

// ---------------------------------------------------------------------------
// Launch
// ---------------------------------------------------------------------------
extern "C" void kernel_launch(void* const* d_in, const int* in_sizes, int n_in,
                              void* d_out, int out_size, void* d_ws, size_t ws_size,
                              hipStream_t stream) {
  const float* x      = (const float*)d_in[0];
  const float* Wqkvf  = (const float*)d_in[1];
  const float* Wconv  = (const float*)d_in[2];
  const float* Wproj  = (const float*)d_in[3];
  const float* Wff1   = (const float*)d_in[4];
  const float* Wff2   = (const float*)d_in[5];
  const float* ln1g   = (const float*)d_in[6];
  const float* ln1b   = (const float*)d_in[7];
  const float* ln2g   = (const float*)d_in[8];
  const float* ln2b   = (const float*)d_in[9];
  float* out = (float*)d_out;

  char* ws = (char*)d_ws;
  size_t off = 0;
  auto alloc = [&](size_t bytes) -> void* {
    void* p = ws + off;
    off = (off + bytes + 255) & ~(size_t)255;
    return p;
  };
  __bf16* bWq   = (__bf16*)alloc((size_t)4 * Dm * Dm * 2);       // (4D, D)
  __bf16* bWp   = (__bf16*)alloc((size_t)Dm * Dm * 2);           // (D, D)
  __bf16* bWf1t = (__bf16*)alloc((size_t)4 * Dm * Dm * 2);       // W_ff1^T
  __bf16* bWf2t = (__bf16*)alloc((size_t)Dm * 4 * Dm * 2);       // W_ff2^T
  __bf16* bWct  = (__bf16*)alloc((size_t)256 * 64 * 2);          // (d'*4+f, c)
  __bf16* h1    = (__bf16*)alloc((size_t)BTm * Dm * 2);
  __bf16* qkvf  = (__bf16*)alloc((size_t)BTm * 4 * Dm * 2);
  __bf16* up    = (__bf16*)alloc((size_t)BTm * Fm * Dm * 2);
  __bf16* kvf   = (__bf16*)alloc((size_t)BTm * Fm * 2 * Dm * 2);
  __bf16* attnb = (__bf16*)alloc((size_t)BTm * Dm * 2);
  float*  x2    = (float*) alloc((size_t)BTm * Dm * 4);
  __bf16* h2    = (__bf16*)alloc((size_t)BTm * Dm * 2);
  __bf16* gbuf  = (__bf16*)alloc((size_t)BTm * 4 * Dm * 2);
  (void)ws_size; (void)in_sizes; (void)n_in; (void)out_size;

  cvt_bf16_kernel<<<dim3(4096), dim3(256), 0, stream>>>(Wqkvf, bWq, 4 * Dm * Dm);
  cvt_bf16_kernel<<<dim3(2048), dim3(256), 0, stream>>>(Wproj, bWp, Dm * Dm);
  cvt_t_bf16_kernel<<<dim3(4096), dim3(256), 0, stream>>>(Wff1, bWf1t, Dm, 4 * Dm);
  cvt_t_bf16_kernel<<<dim3(4096), dim3(256), 0, stream>>>(Wff2, bWf2t, 4 * Dm, Dm);
  cvt_t_bf16_kernel<<<dim3(64),   dim3(256), 0, stream>>>(Wconv, bWct, 64, 256);

  ln_bf16_kernel<<<dim3(BTm), dim3(256), 0, stream>>>(x, ln1g, ln1b, h1);

  // qkvf = h1 @ W_qkvf^T : (2048 x 4096), K=1024
  gemm_wmma_kernel<<<dim3(4 * Dm / 128, BTm / 128), dim3(256), 0, stream>>>(
      h1, Dm, bWq, Dm, Dm, nullptr, nullptr, qkvf, 4 * Dm, 0);

  upfuture_kernel<<<dim3(Tm / 16, Hm, Bm), dim3(32), 0, stream>>>(qkvf, bWct, up);

  // [k_future|v_future] = up @ W_qkvf[D:3D]^T : (8192 x 2048), K=1024
  gemm_wmma_kernel<<<dim3(2 * Dm / 128, BTm * Fm / 128), dim3(256), 0, stream>>>(
      up, Dm, bWq + (size_t)Dm * Dm, Dm, Dm, nullptr, nullptr, kvf, 2 * Dm, 0);

  attn_kernel<<<dim3(Tm / 16, Hm, Bm), dim3(32), 0, stream>>>(qkvf, kvf, attnb);

  // x2 = x + attn @ W_proj^T : (2048 x 1024), K=1024
  gemm_wmma_kernel<<<dim3(Dm / 128, BTm / 128), dim3(256), 0, stream>>>(
      attnb, Dm, bWp, Dm, Dm, x, x2, nullptr, Dm, 0);

  ln_bf16_kernel<<<dim3(BTm), dim3(256), 0, stream>>>(x2, ln2g, ln2b, h2);

  // g = gelu(h2 @ W_ff1) : (2048 x 4096), K=1024
  gemm_wmma_kernel<<<dim3(4 * Dm / 128, BTm / 128), dim3(256), 0, stream>>>(
      h2, Dm, bWf1t, Dm, Dm, nullptr, nullptr, gbuf, 4 * Dm, 1);

  // out = x2 + g @ W_ff2 : (2048 x 1024), K=4096
  gemm_wmma_kernel<<<dim3(Dm / 128, BTm / 128), dim3(256), 0, stream>>>(
      gbuf, 4 * Dm, bWf2t, 4 * Dm, 4 * Dm, x2, out, nullptr, Dm, 0);
}
